// SNN_23510650978445
// MI455X (gfx1250) — matrix operand order repaired and checked by reading, not compiled, not run on hardware
//
#include <hip/hip_runtime.h>

// ---------------------------------------------------------------------------
// SNN forward (2-layer Leaky LIF) for MI455X / gfx1250.
// f16 WMMA (16x16x32, f32 accum) for both GEMMs; f32 state & outputs.
// Layer2 stages its shared A-operand via global_load_async_to_lds_b128.
// Recorded outputs (302 MB write-once) use non-temporal stores to keep the
// 192 MB L2 free for weights / x / state.
// ---------------------------------------------------------------------------

typedef _Float16 v16h __attribute__((ext_vector_type(16)));
typedef _Float16 v8h  __attribute__((ext_vector_type(8)));
typedef float    v8f  __attribute__((ext_vector_type(8)));

union F16Frag { v16h v; v8h h[2]; };

#define BATCH 128
#define NI    1024
#define NH    2048
#define NO    128
#define BETA1 0.9f
#define BETA2 0.8f

// ---------------------------------------------------------------------------
// Init: convert weights to f16 in workspace, zero membrane state.
// Idempotent => deterministic across calls.
// ---------------------------------------------------------------------------
__global__ __launch_bounds__(256) void snn_init(
    const float* __restrict__ W1, const float* __restrict__ W2,
    _Float16* __restrict__ W1h, _Float16* __restrict__ W2h,
    float* __restrict__ mem1, float* __restrict__ mem2) {
  int i = blockIdx.x * blockDim.x + threadIdx.x;
  if (i < NH * NI)    W1h[i]  = (_Float16)W1[i];
  if (i < NO * NH)    W2h[i]  = (_Float16)W2[i];
  if (i < BATCH * NH) mem1[i] = 0.0f;
  if (i < BATCH * NO) mem2[i] = 0.0f;
}

// ---------------------------------------------------------------------------
// Layer 1: cur1 = xt @ W1^T + b1; LIF update; record spk1 (f32 NT + f16 copy).
// Grid: 64 blocks (8 M-tiles x 8 N-groups of 256), 256 threads = 8 waves.
// Each wave owns a 16x32 output tile (2 accumulators, A-frag reuse x2).
// ---------------------------------------------------------------------------
__global__ __launch_bounds__(256) void snn_layer1(
    const float* __restrict__ x,        // (T,B,NI) f32
    const _Float16* __restrict__ W1h,   // (NH,NI)  f16, L2 resident
    const float* __restrict__ b1,
    float* __restrict__ mem1,           // (B,NH) state
    _Float16* __restrict__ spk1h,       // (B,NH) f16 activations for layer 2
    float* __restrict__ spk1_rec,       // (T,B,NH) output
    float* __restrict__ mem1_trace,     // (T,) output
    int t) {
  __shared__ _Float16 sA[16 * NI];      // 32 KB: xt row-block as f16

  const int tid  = threadIdx.x;
  const int lane = tid & 31;
  const int wave = tid >> 5;

  const int mtile = blockIdx.x & 7;     // batch tile (16 rows)
  const int ngrp  = blockIdx.x >> 3;    // 256-wide hidden-neuron group (8 groups)
  const int m0    = mtile * 16;
  const int nbase = ngrp * 256 + wave * 32;

  // ---- Stage xt[m0:m0+16, :] into LDS, converting f32 -> f16 ----
  const float4* xt4 = (const float4*)(x + (size_t)t * BATCH * NI);
  for (int i = tid; i < 16 * (NI / 4); i += 256) {
    int row = i >> 8;                   // NI/4 = 256 float4 per row
    int c4  = i & 255;
    float4 v = xt4[(size_t)(m0 + row) * (NI / 4) + c4];
    int o = row * NI + c4 * 4;
    sA[o + 0] = (_Float16)v.x;
    sA[o + 1] = (_Float16)v.y;
    sA[o + 2] = (_Float16)v.z;
    sA[o + 3] = (_Float16)v.w;
  }
  __syncthreads();

  // WMMA 16-bit A layout: lanes 0-15 -> row m=lane, K {0..7}U{16..23};
  // lanes 16-31 -> K {8..15}U{24..31}.
  const int arow  = lane & 15;
  const int akoff = (lane >> 4) << 3;   // 0 or 8
  // 16-bit B layout: lane -> col n=lane&15, K same as A: {0..7}U{16..23} / {8..15}U{24..31}.
  const int bcol  = lane & 15;
  const int bkoff = (lane >> 4) << 3;   // 0 or 8

  v8f acc[2] = {};
  const _Float16* aBase = &sA[arow * NI];

  for (int k0 = 0; k0 < NI; k0 += 32) {
    F16Frag a;
    const _Float16* ap = aBase + k0 + akoff;
    a.h[0] = *(const v8h*)(ap);         // ds_load_b128
    a.h[1] = *(const v8h*)(ap + 16);
#pragma unroll
    for (int j = 0; j < 2; ++j) {
      const _Float16* bp = &W1h[(size_t)(nbase + j * 16 + bcol) * NI + k0 + bkoff];
      F16Frag b;
      b.h[0] = *(const v8h*)(bp);       // global_load_b128 (L2 hit)
      b.h[1] = *(const v8h*)(bp + 16);
      acc[j] = __builtin_amdgcn_wmma_f32_16x16x32_f16(
          false, a.v, false, b.v, (short)0, acc[j], false, false);
    }
  }

  // ---- Fused LIF epilogue. C layout: lane l, VGPR v -> m = 8*(l>=16)+v, n = l&15.
  const int mloc = (lane >> 4) << 3;
#pragma unroll
  for (int j = 0; j < 2; ++j) {
    int n = nbase + j * 16 + bcol;
    float bias = b1[n];
#pragma unroll
    for (int v = 0; v < 8; ++v) {
      int m = m0 + mloc + v;
      size_t idx = (size_t)m * NH + n;
      float old   = mem1[idx];
      float cur   = acc[j][v] + bias;
      float reset = (old > 1.0f) ? 1.0f : 0.0f;   // heaviside(mem - 1), pre-update
      float nm    = BETA1 * old + cur - reset;
      float spk   = (nm > 1.0f) ? 1.0f : 0.0f;
      mem1[idx]  = nm;
      spk1h[idx] = (_Float16)spk;                 // exact in f16 ({0,1})
      __builtin_nontemporal_store(spk, &spk1_rec[(size_t)t * BATCH * NH + idx]);
      if (idx == 0) mem1_trace[t] = nm;           // mem1[0,0] trace
    }
  }
}

// ---------------------------------------------------------------------------
// Layer 2: cur2 = spk1 @ W2^T + b2; LIF update; record spk2 + mem2 (NT).
// Grid: 8 blocks (one batch tile each) x 8 waves (16 output cols each).
// The 16x2048 f16 A-block (64 KB) is staged into dynamic LDS once per block
// with async global->LDS copies (ASYNCcnt), then all 8 waves read fragments
// via ds_load_b128.
// ---------------------------------------------------------------------------
__global__ __launch_bounds__(256) void snn_layer2(
    const _Float16* __restrict__ spk1h, // (B,NH) f16
    const _Float16* __restrict__ W2h,   // (NO,NH) f16, L2 resident
    const float* __restrict__ b2,
    float* __restrict__ mem2,           // (B,NO) state
    float* __restrict__ spk2_rec,       // (T,B,NO) output
    float* __restrict__ mem2_rec,       // (T,B,NO) output
    int t) {
  extern __shared__ _Float16 sA2[];     // 16 * NH f16 = 64 KB (dynamic LDS)

  const int tid   = threadIdx.x;
  const int lane  = tid & 31;
  const int wave  = tid >> 5;
  const int m0    = blockIdx.x * 16;
  const int ntile = wave * 16;

  // ---- Async-stage spk1h[m0:m0+16, :] into LDS (16B chunks, 4096 total) ----
  for (int i = tid; i < 16 * NH / 8; i += 256) {
    unsigned int lds_off = (unsigned int)(uintptr_t)(&sA2[i * 8]);
    const void* gp = (const void*)(spk1h + (size_t)(m0 + (i >> 8)) * NH + (i & 255) * 8);
    asm volatile("global_load_async_to_lds_b128 %0, %1, off"
                 :: "v"(lds_off), "v"(gp)
                 : "memory");
  }
  asm volatile("s_wait_asynccnt 0" ::: "memory");
  __syncthreads();

  const int arow  = lane & 15;
  const int akoff = (lane >> 4) << 3;
  const int bcol  = lane & 15;
  const int bkoff = (lane >> 4) << 3;

  v8f acc = {};
  const _Float16* aBase = &sA2[arow * NH];
  const _Float16* bRow  = W2h + (size_t)(ntile + bcol) * NH;

  for (int k0 = 0; k0 < NH; k0 += 32) {
    F16Frag a, b;
    const _Float16* ap = aBase + k0 + akoff;
    const _Float16* bp = bRow + k0 + bkoff;
    a.h[0] = *(const v8h*)(ap);         // ds_load_b128
    a.h[1] = *(const v8h*)(ap + 16);
    b.h[0] = *(const v8h*)(bp);
    b.h[1] = *(const v8h*)(bp + 16);
    acc = __builtin_amdgcn_wmma_f32_16x16x32_f16(
        false, a.v, false, b.v, (short)0, acc, false, false);
  }

  int n = ntile + bcol;
  float bias = b2[n];
  const int mloc = (lane >> 4) << 3;
#pragma unroll
  for (int v = 0; v < 8; ++v) {
    int m = m0 + mloc + v;
    size_t idx = (size_t)m * NO + n;
    float old   = mem2[idx];
    float cur   = acc[v] + bias;
    float reset = (old > 1.0f) ? 1.0f : 0.0f;
    float nm    = BETA2 * old + cur - reset;
    float spk   = (nm > 1.0f) ? 1.0f : 0.0f;
    mem2[idx] = nm;
    size_t ridx = (size_t)t * BATCH * NO + idx;
    __builtin_nontemporal_store(spk, &spk2_rec[ridx]);
    __builtin_nontemporal_store(nm,  &mem2_rec[ridx]);
  }
}

// ---------------------------------------------------------------------------
extern "C" void kernel_launch(void* const* d_in, const int* in_sizes, int n_in,
                              void* d_out, int out_size, void* d_ws, size_t ws_size,
                              hipStream_t stream) {
  const float* x  = (const float*)d_in[0];  // (T,B,NI)
  const float* W1 = (const float*)d_in[1];  // (NH,NI)
  const float* b1 = (const float*)d_in[2];
  const float* W2 = (const float*)d_in[3];  // (NO,NH)
  const float* b2 = (const float*)d_in[4];
  const int T = in_sizes[0] / (BATCH * NI);

  // Workspace layout (~6.1 MB total)
  char* ws = (char*)d_ws;
  _Float16* W1h   = (_Float16*)(ws);                              // 4 MB
  _Float16* W2h   = (_Float16*)(ws + (4u << 20));                 // 512 KB
  _Float16* spk1h = (_Float16*)(ws + (4u << 20) + (512u << 10));  // 512 KB
  float*    mem1  = (float*)   (ws + (5u << 20));                 // 1 MB
  float*    mem2  = (float*)   (ws + (6u << 20));                 // 64 KB

  // Output layout: spk1_rec | mem1_trace | spk2_rec | mem2_rec (flat, in order)
  float* out        = (float*)d_out;
  float* spk1_rec   = out;
  float* mem1_trace = spk1_rec + (size_t)T * BATCH * NH;
  float* spk2_rec   = mem1_trace + T;
  float* mem2_rec   = spk2_rec + (size_t)T * BATCH * NO;

  snn_init<<<(NH * NI + 255) / 256, 256, 0, stream>>>(W1, W2, W1h, W2h, mem1, mem2);

  const size_t lds2 = (size_t)16 * NH * sizeof(_Float16);  // 64 KB dynamic LDS
  for (int t = 0; t < T; ++t) {
    snn_layer1<<<64, 256, 0, stream>>>(x, W1h, b1, mem1, spk1h,
                                       spk1_rec, mem1_trace, t);
    snn_layer2<<<8, 256, lds2, stream>>>(spk1h, W2h, b2, mem2,
                                         spk2_rec, mem2_rec, t);
  }
}
